// VectorQuantizer_71708773974878
// MI455X (gfx1250) — matrix operand order repaired and checked by reading, not compiled, not run on hardware
//
#include <hip/hip_runtime.h>
#include <hip/hip_bf16.h>

#define BATCH 4096
#define EDIM  256
#define NE    4096
#define SK_ITERS 100

typedef __attribute__((ext_vector_type(16))) __bf16 v16bf;
typedef __attribute__((ext_vector_type(8)))  __bf16 v8bf;
typedef __attribute__((ext_vector_type(8)))  float  v8f;

// ---------------- workspace layout (float-unit offsets) ----------------
static constexpr size_t OFF_M    = 0;                       // 4096*4096 f32 kernel matrix M
static constexpr size_t OFF_XH   = (size_t)NE * NE;         // bf16 hi of x   (1048576 u16 = 524288 f32)
static constexpr size_t OFF_XL   = OFF_XH + 524288;
static constexpr size_t OFF_CH   = OFF_XL + 524288;
static constexpr size_t OFF_CL   = OFF_CH + 524288;
static constexpr size_t OFF_XN   = OFF_CL + 524288;         // ||x_i||^2        [4096]
static constexpr size_t OFF_CN   = OFF_XN + NE;             // ||c_j||^2        [4096]
static constexpr size_t OFF_T    = OFF_CN + NE;             // col accum t      [4096]
static constexpr size_t OFF_V    = OFF_T  + NE;             // v scaling        [4096]
static constexpr size_t OFF_S    = OFF_V  + NE;             // row accum s      [4096]
static constexpr size_t OFF_U    = OFF_S  + NE;             // u scaling        [4096]
static constexpr size_t OFF_KEYS = OFF_U  + NE;             // 2 uint min/max keys
static constexpr size_t OFF_SCAL = OFF_KEYS + 2;            // mn, 1/(amp*eps)
static constexpr size_t OFF_SSE  = OFF_SCAL + 2;            // sum of squared errors
static constexpr size_t OFF_IDX  = OFF_SSE + 1;             // int indices [4096]

// ---------------- helpers ----------------
static __device__ __forceinline__ unsigned short f2bf(float f) {
  unsigned u = __float_as_uint(f);
  unsigned r = u + 0x7FFFu + ((u >> 16) & 1u);   // round-to-nearest-even
  return (unsigned short)(r >> 16);
}
static __device__ __forceinline__ float bf2f(unsigned short h) {
  return __uint_as_float(((unsigned)h) << 16);
}
// monotone unsigned key for float atomics min/max
static __device__ __forceinline__ unsigned fkey(float f) {
  unsigned u = __float_as_uint(f);
  return (u & 0x80000000u) ? ~u : (u | 0x80000000u);
}
static __device__ __forceinline__ float funkey(unsigned k) {
  unsigned u = (k & 0x80000000u) ? (k & 0x7FFFFFFFu) : ~k;
  return __uint_as_float(u);
}

// ---------------- kernels ----------------
__global__ void vq_init(float* W) {
  int i = blockIdx.x * blockDim.x + threadIdx.x;
  if (i < NE) {
    (W + OFF_U)[i] = 1.0f;
    (W + OFF_T)[i] = 0.0f;
    (W + OFF_S)[i] = 0.0f;
  }
  if (i == 0) {
    unsigned* keys = (unsigned*)(W + OFF_KEYS);
    keys[0] = 0xFFFFFFFFu;   // min key
    keys[1] = 0u;            // max key
    (W + OFF_SSE)[0] = 0.0f;
  }
}

// split f32 -> (bf16 hi, bf16 lo) for both x and codebook
__global__ void vq_convert(const float* __restrict__ X, const float* __restrict__ CB, float* W) {
  unsigned short* XH = (unsigned short*)(W + OFF_XH);
  unsigned short* XL = (unsigned short*)(W + OFF_XL);
  unsigned short* CH = (unsigned short*)(W + OFF_CH);
  unsigned short* CL = (unsigned short*)(W + OFF_CL);
  int stride = gridDim.x * blockDim.x;
  for (int i = blockIdx.x * blockDim.x + threadIdx.x; i < BATCH * EDIM; i += stride) {
    float x = X[i];
    unsigned short xh = f2bf(x);
    XH[i] = xh; XL[i] = f2bf(x - bf2f(xh));
    float c = CB[i];
    unsigned short ch = f2bf(c);
    CH[i] = ch; CL[i] = f2bf(c - bf2f(ch));
  }
}

// one wave per row: squared L2 norms of x rows and codebook rows
__global__ void vq_norms(const float* __restrict__ X, const float* __restrict__ CB, float* W) {
  int lane = threadIdx.x & 31, wid = threadIdx.x >> 5;
  int w = blockIdx.x * 8 + wid;
  const float* src; float* dst; int row;
  if (w < NE) { src = X;  dst = W + OFF_XN; row = w; }
  else        { src = CB; dst = W + OFF_CN; row = w - NE; }
  float acc = 0.0f;
  const float* p = src + (size_t)row * EDIM;
  #pragma unroll
  for (int k = lane; k < EDIM; k += 32) { float v = p[k]; acc += v * v; }
  for (int off = 16; off; off >>= 1) acc += __shfl_xor(acc, off, 32);
  if (lane == 0) dst[row] = acc;
}

// WMMA GEMM: d[i,j] = ||x_i||^2 + ||c_j||^2 - 2 * x_i . c_j
// split-bf16: x.c ~= xh.ch + xh.cl + xl.ch  (3 WMMAs per K=32 chunk)
__global__ void vq_gemm(float* W, float* __restrict__ D) {
  const unsigned short* XH = (const unsigned short*)(W + OFF_XH);
  const unsigned short* XL = (const unsigned short*)(W + OFF_XL);
  const unsigned short* CH = (const unsigned short*)(W + OFF_CH);
  const unsigned short* CL = (const unsigned short*)(W + OFF_CL);
  const float* XN = W + OFF_XN;
  const float* CN = W + OFF_CN;
  unsigned* keys = (unsigned*)(W + OFF_KEYS);

  int lane = threadIdx.x & 31, wid = threadIdx.x >> 5;
  int tile = blockIdx.x * 8 + wid;      // 65536 tiles total
  int ti = tile >> 8;                   // 16-row tile index
  int tj = tile & 255;                  // 16-col tile index
  int half = lane >> 4;                 // lane half selects K sub-ranges
  int m16  = lane & 15;

  // A fragment source row (x) and B fragment source row (codebook, acting as B column)
  const int arow = ti * 16 + m16;
  const int bcol = tj * 16 + m16;

  v8f acc = {};
  #pragma unroll
  for (int kk = 0; kk < EDIM; kk += 32) {
    // A layout (16-bit, 16x32): lanes<16 hold K {kk..kk+7, kk+16..kk+23}; lanes>=16 shifted by 8
    const unsigned short* ah = XH + (size_t)arow * EDIM + kk + half * 8;
    const unsigned short* al = XL + (size_t)arow * EDIM + kk + half * 8;
    v8bf ah0 = *(const v8bf*)ah;
    v8bf ah1 = *(const v8bf*)(ah + 16);
    v8bf al0 = *(const v8bf*)al;
    v8bf al1 = *(const v8bf*)(al + 16);
    v16bf Ah = __builtin_shufflevector(ah0, ah1, 0,1,2,3,4,5,6,7,8,9,10,11,12,13,14,15);
    v16bf Al = __builtin_shufflevector(al0, al1, 0,1,2,3,4,5,6,7,8,9,10,11,12,13,14,15);
    // B layout (16-bit, 32x16): lane n=lane&15, lanes<16 hold K {kk..kk+7, kk+16..kk+23}; lanes>=16 shifted by 8
    const unsigned short* bh = CH + (size_t)bcol * EDIM + kk + half * 8;
    const unsigned short* bl = CL + (size_t)bcol * EDIM + kk + half * 8;
    v8bf bh0 = *(const v8bf*)bh;
    v8bf bh1 = *(const v8bf*)(bh + 16);
    v8bf bl0 = *(const v8bf*)bl;
    v8bf bl1 = *(const v8bf*)(bl + 16);
    v16bf Bh = __builtin_shufflevector(bh0, bh1, 0,1,2,3,4,5,6,7,8,9,10,11,12,13,14,15);
    v16bf Bl = __builtin_shufflevector(bl0, bl1, 0,1,2,3,4,5,6,7,8,9,10,11,12,13,14,15);

    acc = __builtin_amdgcn_wmma_f32_16x16x32_bf16(false, Ah, false, Bh, (short)0, acc, false, false);
    acc = __builtin_amdgcn_wmma_f32_16x16x32_bf16(false, Ah, false, Bl, (short)0, acc, false, false);
    acc = __builtin_amdgcn_wmma_f32_16x16x32_bf16(false, Al, false, Bh, (short)0, acc, false, false);
  }

  // C/D layout: lane covers column n = lane&15; VGPR r holds row r (+8 for upper lane half)
  int n = tj * 16 + m16;
  float cn = CN[n];
  float lmin = 3.4e38f, lmax = -3.4e38f;
  #pragma unroll
  for (int r = 0; r < 8; ++r) {
    int mrow = ti * 16 + r + half * 8;
    float dv = XN[mrow] + cn - 2.0f * acc[r];
    D[(size_t)mrow * NE + n] = dv;
    lmin = fminf(lmin, dv);
    lmax = fmaxf(lmax, dv);
  }
  for (int off = 16; off; off >>= 1) {
    lmin = fminf(lmin, __shfl_xor(lmin, off, 32));
    lmax = fmaxf(lmax, __shfl_xor(lmax, off, 32));
  }
  if (lane == 0) {
    atomicMin(&keys[0], fkey(lmin));
    atomicMax(&keys[1], fkey(lmax));
  }
}

// derive centering scalars: mn and 1/(amp*eps)
__global__ void vq_scal(float* W) {
  unsigned* keys = (unsigned*)(W + OFF_KEYS);
  float mn = funkey(keys[0]);
  float mx = funkey(keys[1]);
  float middle = 0.5f * (mx + mn);
  float amp = mx - middle + 1e-5f;
  (W + OFF_SCAL)[0] = mn;
  (W + OFF_SCAL)[1] = 1.0f / (amp * 0.05f);
}

// M = exp(-(d - mn)/(amp*eps))  -- 64MB, lives in L2 for the whole Sinkhorn loop
__global__ void vq_exp(const float* __restrict__ D, float* W) {
  float mn = (W + OFF_SCAL)[0];
  float a  = (W + OFF_SCAL)[1];
  float* Mm = W + OFF_M;
  size_t stride = (size_t)gridDim.x * blockDim.x;
  for (size_t i = (size_t)blockIdx.x * blockDim.x + threadIdx.x; i < (size_t)BATCH * NE; i += stride)
    Mm[i] = __expf(-(D[i] - mn) * a);
}

// t_j += sum_i M[i,j] * u[i]   (coalesced down columns, split-rows via atomics)
__global__ void vq_colsum(float* W) {
  const float* Mm = W + OFF_M;
  const float* U  = W + OFF_U;
  float* T = W + OFF_T;
  int j  = blockIdx.x * 256 + threadIdx.x;
  int r0 = blockIdx.y * 128;
  float acc = 0.0f;
  const float* p = Mm + (size_t)r0 * NE + j;
  #pragma unroll 4
  for (int r = 0; r < 128; ++r) acc += p[(size_t)r * NE] * U[r0 + r];
  atomicAdd(&T[j], acc);
}

__global__ void vq_vupd(float* W) {
  int j = blockIdx.x * blockDim.x + threadIdx.x;
  if (j < NE) {
    (W + OFF_V)[j] = 1.0f / ((float)NE * (W + OFF_T)[j]);
    (W + OFF_S)[j] = 0.0f;
  }
}

// s_i += sum_j M[i,j] * v[j]   (one wave per row, column slabs)
__global__ void vq_rowsum(float* W) {
  const float* Mm = W + OFF_M;
  const float* V  = W + OFF_V;
  float* S = W + OFF_S;
  int lane = threadIdx.x & 31, wid = threadIdx.x >> 5;
  int i  = blockIdx.x * 8 + wid;
  int c0 = blockIdx.y * 512;
  const float* row = Mm + (size_t)i * NE;
  float acc = 0.0f;
  #pragma unroll 4
  for (int c = c0 + lane; c < c0 + 512; c += 32) acc += row[c] * V[c];
  for (int off = 16; off; off >>= 1) acc += __shfl_xor(acc, off, 32);
  if (lane == 0) atomicAdd(&S[i], acc);
}

__global__ void vq_uupd(float* W) {
  int i = blockIdx.x * blockDim.x + threadIdx.x;
  if (i < NE) {
    (W + OFF_U)[i] = 1.0f / ((float)BATCH * (W + OFF_S)[i]);
    (W + OFF_T)[i] = 0.0f;
  }
}

// argmax_j u_i*M[i,j]*v[j] == argmax_j M[i,j]*v[j]; one wave per row
__global__ void vq_argmax(float* W, float* __restrict__ out_idx_f) {
  const float* Mm = W + OFF_M;
  const float* V  = W + OFF_V;
  int* IDX = (int*)(W + OFF_IDX);
  int lane = threadIdx.x & 31, wid = threadIdx.x >> 5;
  int i = blockIdx.x * 8 + wid;
  const float* row = Mm + (size_t)i * NE;
  float best = -3.4e38f; int bj = 0;
  for (int c = lane; c < NE; c += 32) {
    float q = row[c] * V[c];
    if (q > best) { best = q; bj = c; }
  }
  for (int off = 16; off; off >>= 1) {
    float ob = __shfl_xor(best, off, 32);
    int   oj = __shfl_xor(bj,   off, 32);
    if (ob > best || (ob == best && oj < bj)) { best = ob; bj = oj; }
  }
  if (lane == 0) { IDX[i] = bj; out_idx_f[i] = (float)bj; }
}

// x_q_st (== x_q forward) + accumulate SSE for the loss
__global__ void vq_gather(const float* __restrict__ X, const float* __restrict__ CB,
                          float* W, float* __restrict__ xq) {
  const int* IDX = (const int*)(W + OFF_IDX);
  int lane = threadIdx.x & 31;
  int stride = gridDim.x * blockDim.x;
  float sse = 0.0f;
  for (int i = blockIdx.x * blockDim.x + threadIdx.x; i < BATCH * EDIM; i += stride) {
    int row = i >> 8, k = i & 255;
    float cb = CB[(size_t)IDX[row] * EDIM + k];
    xq[i] = cb;
    float dlt = cb - X[i];
    sse += dlt * dlt;
  }
  for (int off = 16; off; off >>= 1) sse += __shfl_xor(sse, off, 32);
  if (lane == 0) atomicAdd(W + OFF_SSE, sse);
}

__global__ void vq_loss(const float* __restrict__ W, float* __restrict__ loss_out) {
  // loss = codebook_loss + beta*commitment = (1 + 0.25) * mean((xq-x)^2)
  loss_out[0] = 1.25f * W[OFF_SSE] / (float)(BATCH * EDIM);
}

// ---------------- launch ----------------
extern "C" void kernel_launch(void* const* d_in, const int* in_sizes, int n_in,
                              void* d_out, int out_size, void* d_ws, size_t ws_size,
                              hipStream_t stream) {
  (void)in_sizes; (void)n_in; (void)out_size; (void)ws_size;
  const float* X  = (const float*)d_in[0];   // [4096, 256]
  const float* CB = (const float*)d_in[1];   // [4096, 256]
  float* out = (float*)d_out;
  float* W   = (float*)d_ws;

  float* out_xq   = out;                         // 4096*256
  float* out_loss = out + (size_t)BATCH * EDIM;  // 1
  float* out_idx  = out_loss + 1;                // 4096
  float* out_d    = out_idx + NE;                // 4096*4096

  vq_init   <<<16, 256, 0, stream>>>(W);
  vq_convert<<<1024, 256, 0, stream>>>(X, CB, W);
  vq_norms  <<<1024, 256, 0, stream>>>(X, CB, W);          // 8192 waves: 4096 x-rows + 4096 cb-rows
  vq_gemm   <<<8192, 256, 0, stream>>>(W, out_d);          // 65536 16x16 tiles, 8 waves/block
  vq_scal   <<<1, 1, 0, stream>>>(W);
  vq_exp    <<<4096, 256, 0, stream>>>(out_d, W);

  for (int it = 0; it < SK_ITERS; ++it) {
    vq_colsum<<<dim3(16, 32), 256, 0, stream>>>(W);
    vq_vupd  <<<16, 256, 0, stream>>>(W);
    vq_rowsum<<<dim3(512, 8), 256, 0, stream>>>(W);
    vq_uupd  <<<16, 256, 0, stream>>>(W);
  }

  vq_argmax<<<512, 256, 0, stream>>>(W, out_idx);
  vq_gather<<<1024, 256, 0, stream>>>(X, CB, W, out_xq);
  vq_loss  <<<1, 1, 0, stream>>>(W, out_loss);
}
